// RNN_62912680952256
// MI455X (gfx1250) — matrix operand order repaired and checked by reading, not compiled, not run on hardware
//
#include <hip/hip_runtime.h>
#include <math.h>
#include <stdint.h>

typedef unsigned short u16;
typedef unsigned int   u32;
typedef __attribute__((ext_vector_type(16))) __bf16 v16bf;
typedef __attribute__((ext_vector_type(8)))  float  v8f;
typedef __attribute__((ext_vector_type(4)))  unsigned int u32x4;
typedef __attribute__((ext_vector_type(8)))  int i32x8;
typedef __attribute__((ext_vector_type(4)))  int i32x4;

struct alignas(16) U128 { u32 x, y, z, w; };
struct alignas(32) U256 { U128 lo, hi; };

#define RNN_T    512
#define RNN_B    64
#define RNN_I    512
#define RNN_H    1024
#define RNN_NBLK 32   // persistent recurrence blocks (32 blk x 8 waves = 256 tiles)

#if defined(__HIP_DEVICE_COMPILE__) && __has_builtin(__builtin_amdgcn_tensor_load_to_lds)
#define RNN_HAS_TDM 1
#else
#define RNN_HAS_TDM 0
#endif

// Build a v16bf WMMA fragment from two 16-byte chunks of bf16 data.
// Works for both global and LDS pointers (forceinline -> addrspace inference
// turns the LDS case into ds_load_b128 pairs).
template <typename P>
static __device__ __forceinline__ v16bf frag_from(P p0, P p1) {
  U256 t;
  t.lo = *(const U128*)p0;
  t.hi = *(const U128*)p1;
  return __builtin_bit_cast(v16bf, t);
}

// fp32 -> bf16, round-to-nearest-even (bit trick).
static __device__ __forceinline__ u16 f32_to_bf16(float f) {
  u32 u = __builtin_bit_cast(u32, f);
  u32 r = u + 0x7FFFu + ((u >> 16) & 1u);
  return (u16)(r >> 16);
}

#if RNN_HAS_TDM
// TDM: DMA n_elems bf16 from global to LDS as a 2-D tile of <=32768-elem rows
// (tile_dim0 is a 16-bit field). Descriptor per CDNA5 ISA 8.3/8.4:
//   group0: count=1 | lds_addr | global_addr[56:0] | type=2
//   group1: data_size=1(2B) | tensor_dim0/1 | tile_dim0/1 | dim0_stride
// Groups 2/3 (3D+/gather) unused -> zeros. 6-arg builtin (clang-23 form).
// Issued from one wave (EXEC ignored, operands are SGPRs); completion via
// s_wait_tensorcnt 0, visibility to other waves via __syncthreads().
static __device__ __forceinline__ void tdm_load_u16(void* lds_dst,
                                                    const u16* gsrc,
                                                    u32 n_elems) {
  const u32 lds_off = (u32)(unsigned long long)lds_dst;  // low 32b = LDS offset
  const unsigned long long ga = (unsigned long long)gsrc;
  const u32 rows = (n_elems > 32768u) ? (n_elems >> 15) : 1u;
  const u32 cols = (rows > 1u) ? 32768u : n_elems;

  u32x4 g0;
  g0.x = 1u;                                   // count=1, user descriptor
  g0.y = lds_off;                              // lds_addr (bytes)
  g0.z = (u32)ga;                              // global_addr[31:0]
  g0.w = (u32)(ga >> 32) | (2u << 30);         // global_addr[56:32] | type=2

  i32x8 g1;
  g1[0] = (int)(1u << 16);                     // data_size=1 (2 bytes), mask=0
  g1[1] = (int)((cols & 0xFFFFu) << 16);       // tensor_dim0[15:0]
  g1[2] = (int)((cols >> 16) | ((rows & 0xFFFFu) << 16)); // tdim0 hi | tdim1 lo
  g1[3] = (int)((cols & 0xFFFFu) << 16);       // tdim1 hi(=0) | tile_dim0
  g1[4] = (int)rows;                           // tile_dim1 | tile_dim2=0
  g1[5] = (int)cols;                           // tensor_dim0_stride[31:0]
  g1[6] = 0;                                   // stride0 hi | stride1 lo
  g1[7] = 0;                                   // stride1 hi
  i32x4 z4 = {0, 0, 0, 0};
  i32x8 z8 = {0, 0, 0, 0, 0, 0, 0, 0};
  __builtin_amdgcn_tensor_load_to_lds(g0, g1, z4, z4, z8, 0);
}
#endif

__global__ void zero_u32_kernel(u32* __restrict__ p, int n) {
  int i = blockIdx.x * blockDim.x + threadIdx.x;
  if (i < n) p[i] = 0u;
}

__global__ void cvt_f32_to_bf16_kernel(const float* __restrict__ src,
                                       u16* __restrict__ dst, int n) {
  int i = blockIdx.x * blockDim.x + threadIdx.x;
  if (i < n) dst[i] = f32_to_bf16(src[i]);
}

// out[M,N] = A[M,K](bf16) * W[N,K]^T(bf16) + b1[N] + b2[N]   (fp32 accumulate)
// Block = 8 waves sharing one 64-column B-group staged in LDS via TDM.
// Each wave computes a 16(M) x 64(N) macro-tile; A streams from global/L2,
// B comes from LDS (ds_load_b128).
__global__ __launch_bounds__(256) void wmma_xproj_kernel(
    const u16* __restrict__ A, const u16* __restrict__ W,
    const float* __restrict__ b1, const float* __restrict__ b2,
    float* __restrict__ out, int M, int N, int K) {
  extern __shared__ u16 smemB[];              // [64][K] bf16

  const int lane = threadIdx.x & 31;
  const int wave = threadIdx.x >> 5;
  const int l15  = lane & 15;
  const bool hi  = lane >= 16;
  const int ngrp = N >> 6;                    // # of 64-column groups
  const int nbase  = (blockIdx.x % ngrp) << 6;
  const int mblock = blockIdx.x / ngrp;       // 8 M-tiles per block
  const int mtile  = mblock * 8 + wave;

  // Stage the 64 x K B-group (rows nbase..nbase+63 of W) into LDS.
#if RNN_HAS_TDM
  if (wave == 0) {
    tdm_load_u16(smemB, W + (size_t)nbase * K, (u32)(64 * K));
    __builtin_amdgcn_s_wait_tensorcnt(0);
  }
#else
  {
    const uint4* src = (const uint4*)(W + (size_t)nbase * K);
    uint4* dst = (uint4*)smemB;
    const int cnt = (64 * K) >> 3;            // uint4 = 8 bf16
    for (int i = threadIdx.x; i < cnt; i += 256) dst[i] = src[i];
  }
#endif
  __syncthreads();

  const int arow   = mtile * 16 + l15;
  const int a_koff = hi ? 16 : 0;             // A: lanes 16-31 hold K {16..31}
  const int b_koff = hi ? 16 : 0;             // B: lanes 16-31 hold K {16..31}

  v8f acc0 = {0,0,0,0,0,0,0,0};
  v8f acc1 = acc0, acc2 = acc0, acc3 = acc0;

  for (int k0 = 0; k0 < K; k0 += 32) {
    const u16* ap = A + (size_t)arow * K + k0 + a_koff;
    v16bf af = frag_from(ap, ap + 8);
    const u16* wp0 = smemB + (size_t)( 0 + l15) * K + k0 + b_koff;
    const u16* wp1 = smemB + (size_t)(16 + l15) * K + k0 + b_koff;
    const u16* wp2 = smemB + (size_t)(32 + l15) * K + k0 + b_koff;
    const u16* wp3 = smemB + (size_t)(48 + l15) * K + k0 + b_koff;
    acc0 = __builtin_amdgcn_wmma_f32_16x16x32_bf16(false, af, false, frag_from(wp0, wp0 + 8), (short)0, acc0, false, false);
    acc1 = __builtin_amdgcn_wmma_f32_16x16x32_bf16(false, af, false, frag_from(wp1, wp1 + 8), (short)0, acc1, false, false);
    acc2 = __builtin_amdgcn_wmma_f32_16x16x32_bf16(false, af, false, frag_from(wp2, wp2 + 8), (short)0, acc2, false, false);
    acc3 = __builtin_amdgcn_wmma_f32_16x16x32_bf16(false, af, false, frag_from(wp3, wp3 + 8), (short)0, acc3, false, false);
  }

  const int mb = mtile * 16 + (hi ? 8 : 0);   // C/D: lane halves hold M 0-7 / 8-15
  v8f accs[4] = {acc0, acc1, acc2, acc3};
#pragma unroll
  for (int j = 0; j < 4; ++j) {
    const int ncol = nbase + j * 16 + l15;
    const float bias = b1[ncol] + b2[ncol];
#pragma unroll
    for (int v = 0; v < 8; ++v)
      out[(size_t)(mb + v) * N + ncol] = accs[j][v] + bias;
  }
}

// Device-wide split barrier over RNN_NBLK blocks using a per-step counter.
static __device__ __forceinline__ void grid_barrier(int* cnt) {
  __syncthreads();
  if (threadIdx.x == 0) {
    __hip_atomic_fetch_add(cnt, 1, __ATOMIC_RELEASE, __HIP_MEMORY_SCOPE_AGENT);
    while (__hip_atomic_load(cnt, __ATOMIC_ACQUIRE, __HIP_MEMORY_SCOPE_AGENT) < RNN_NBLK)
      __builtin_amdgcn_s_sleep(2);
  }
  __syncthreads();
}

// Persistent recurrence: h_{t+1} = tanh(xproj[t] + h_t @ Whh^T).
// 32 blocks x 8 wave32 = 256 waves; each wave owns one 16x16 tile of the
// 64x1024 hidden state. Per wave, its 16x1024 Whh panel is staged in LDS ONCE
// via TDM (8 x 32KB); the block's shared 16 h-rows are staged per step
// (32KB TDM on the sequential critical path) -> inner K-loop is pure LDS+WMMA.
// Total dynamic LDS: 288 KB (CDNA5 WGP has 320 KB).
__global__ __launch_bounds__(256) void wmma_recurrence_kernel(
    const float* __restrict__ xproj,   // [T][B][H] fp32
    const u16*   __restrict__ Whh,     // [H][H] bf16 row-major
    u16*         __restrict__ hbuf,    // [2][B][H] bf16 ping-pong (pre-zeroed)
    float*       __restrict__ ys_f32,  // [T][B][H] or null
    u16*         __restrict__ ys_bf16, // [T][B][H] or null
    float*       __restrict__ hT,      // [B][H]
    int*         __restrict__ counters) { // [T] (pre-zeroed)
  extern __shared__ u16 smem[];        // [8][16][1024] B panels + [16][1024] A panel

  const int lane = threadIdx.x & 31;
  const int wave = threadIdx.x >> 5;
  const int l15  = lane & 15;
  const bool hi  = lane >= 16;
  const int tile = blockIdx.x * 8 + wave;   // 0..255 = 4 Mtiles x 64 Ntiles
  const int tn   = tile & 63;
  const int tm   = tile >> 6;               // same for all 8 waves in a block
  const int ncol = tn * 16 + l15;
  const int mb   = tm * 16 + (hi ? 8 : 0);
  const int a_koff = hi ? 16 : 0;
  const int b_koff = hi ? 16 : 0;

  u16* bpanel = smem + (size_t)wave * 16 * RNN_H;   // this wave's Whh rows
  u16* apanel = smem + (size_t)8 * 16 * RNN_H;      // block-shared h rows

  // Stage this wave's 16x1024 Whh panel (rows tn*16 .. tn*16+15) into LDS once.
#if RNN_HAS_TDM
  tdm_load_u16(bpanel, Whh + (size_t)(tn * 16) * RNN_H, 16 * RNN_H);
  __builtin_amdgcn_s_wait_tensorcnt(0);
#else
  {
    const uint4* src = (const uint4*)(Whh + (size_t)(tn * 16) * RNN_H);
    uint4* dst = (uint4*)bpanel;
    const int cnt = (16 * RNN_H) >> 3;   // 2048 uint4
    for (int i = lane; i < cnt; i += 32) dst[i] = src[i];
  }
#endif
  __syncthreads();

  for (int t = 0; t < RNN_T; ++t) {
    const u16* hsrc = hbuf + (size_t)(t & 1) * RNN_B * RNN_H;
    // Stage the block's 16 h-rows (tm*16 .. tm*16+15) for this step.
#if RNN_HAS_TDM
    if (wave == 0) {
      tdm_load_u16(apanel, hsrc + (size_t)(tm * 16) * RNN_H, 16 * RNN_H);
      __builtin_amdgcn_s_wait_tensorcnt(0);
    }
#else
    {
      const uint4* src = (const uint4*)(hsrc + (size_t)(tm * 16) * RNN_H);
      uint4* dst = (uint4*)apanel;
      const int cnt = (16 * RNN_H) >> 3;
      for (int i = threadIdx.x; i < cnt; i += 256) dst[i] = src[i];
    }
#endif
    __syncthreads();

    const float* xp = xproj + (size_t)t * RNN_B * RNN_H;
    v8f acc;
#pragma unroll
    for (int v = 0; v < 8; ++v)
      acc[v] = xp[(size_t)(mb + v) * RNN_H + ncol];

#pragma unroll 4
    for (int k0 = 0; k0 < RNN_H; k0 += 32) {
      const u16* ap = apanel + (size_t)l15 * RNN_H + k0 + a_koff;
      v16bf af = frag_from(ap, ap + 8);
      const u16* wp = bpanel + (size_t)l15 * RNN_H + k0 + b_koff;
      v16bf bf = frag_from(wp, wp + 8);
      acc = __builtin_amdgcn_wmma_f32_16x16x32_bf16(false, af, false, bf, (short)0, acc, false, false);
    }

    u16* hdst = hbuf + (size_t)((t + 1) & 1) * RNN_B * RNN_H;
#pragma unroll
    for (int v = 0; v < 8; ++v) {
      const float val = tanhf(acc[v]);
      const size_t idx = (size_t)(mb + v) * RNN_H + ncol;
      hdst[idx] = f32_to_bf16(val);
      if (ys_bf16) ys_bf16[(size_t)t * RNN_B * RNN_H + idx] = f32_to_bf16(val);
      if (ys_f32)  ys_f32 [(size_t)t * RNN_B * RNN_H + idx] = val;
      if (t == RNN_T - 1) hT[idx] = val;
    }
    // Barrier orders: all LDS apanel reads done (safe to restage), all hdst
    // writes visible device-wide before the next step reads them.
    grid_barrier(counters + t);
  }
}

extern "C" void kernel_launch(void* const* d_in, const int* in_sizes, int n_in,
                              void* d_out, int out_size, void* d_ws, size_t ws_size,
                              hipStream_t stream) {
  (void)in_sizes; (void)n_in; (void)out_size; (void)ws_size;
  constexpr int T = RNN_T, B = RNN_B, I = RNN_I, H = RNN_H;

  const float* X    = (const float*)d_in[0];
  const float* Wih0 = (const float*)d_in[1];
  const float* Whh0 = (const float*)d_in[2];
  const float* bih0 = (const float*)d_in[3];
  const float* bhh0 = (const float*)d_in[4];
  const float* Wih1 = (const float*)d_in[5];
  const float* Whh1 = (const float*)d_in[6];
  const float* bih1 = (const float*)d_in[7];
  const float* bhh1 = (const float*)d_in[8];

  // Allow >64KB dynamic LDS (CDNA5 WGP: up to 320KB per workgroup).
  (void)hipFuncSetAttribute((const void*)wmma_xproj_kernel,
                            hipFuncAttributeMaxDynamicSharedMemorySize, 131072);
  (void)hipFuncSetAttribute((const void*)wmma_recurrence_kernel,
                            hipFuncAttributeMaxDynamicSharedMemorySize, 294912);

  // Workspace carve-out (~232 MB total)
  char* ws = (char*)d_ws;
  auto carve = [&](size_t bytes) -> char* {
    char* p = ws; ws += (bytes + 255) & ~(size_t)255; return p;
  };
  u16*   Xbf    = (u16*)carve((size_t)T * B * I * sizeof(u16));     // 32 MB
  u16*   Wih0bf = (u16*)carve((size_t)H * I * sizeof(u16));         // 1 MB
  u16*   Whh0bf = (u16*)carve((size_t)H * H * sizeof(u16));         // 2 MB
  u16*   Wih1bf = (u16*)carve((size_t)H * H * sizeof(u16));         // 2 MB
  u16*   Whh1bf = (u16*)carve((size_t)H * H * sizeof(u16));         // 2 MB
  u16*   ys0bf  = (u16*)carve((size_t)T * B * H * sizeof(u16));     // 64 MB
  float* xproj  = (float*)carve((size_t)T * B * H * sizeof(float)); // 128 MB (reused)
  u16*   hbuf0  = (u16*)carve((size_t)2 * B * H * sizeof(u16));
  u16*   hbuf1  = (u16*)carve((size_t)2 * B * H * sizeof(u16));
  int*   cnts   = (int*)carve((size_t)2 * T * sizeof(int));

  float* ys1 = (float*)d_out;                       // [T][B][H]
  float* h0T = ys1 + (size_t)T * B * H;             // [B][H]
  float* h1T = h0T + (size_t)B * H;                 // [B][H]

  // Zero h ping-pong state and barrier counters (fresh every launch/replay).
  const int hbufU32 = B * H;  // 2*B*H u16 == B*H u32
  zero_u32_kernel<<<(hbufU32 + 255) / 256, 256, 0, stream>>>((u32*)hbuf0, hbufU32);
  zero_u32_kernel<<<(hbufU32 + 255) / 256, 256, 0, stream>>>((u32*)hbuf1, hbufU32);
  zero_u32_kernel<<<(2 * T + 255) / 256, 256, 0, stream>>>((u32*)cnts, 2 * T);

  // fp32 -> bf16 conversions
  auto cvt = [&](const float* s, u16* d, int n) {
    cvt_f32_to_bf16_kernel<<<(n + 255) / 256, 256, 0, stream>>>(s, d, n);
  };
  cvt(X,    Xbf,    T * B * I);
  cvt(Wih0, Wih0bf, H * I);
  cvt(Whh0, Whh0bf, H * H);
  cvt(Wih1, Wih1bf, H * H);
  cvt(Whh1, Whh1bf, H * H);

  // Layer 0: xproj0 = X @ Wih0^T + bih0 + bhh0, then recurrence -> ys0 (bf16), h0T
  {
    const int M = T * B, N = H, K = I;
    const int blocks = (M / 128) * (N / 64);   // 8 M-tiles per block
    wmma_xproj_kernel<<<blocks, 256, (size_t)64 * K * sizeof(u16), stream>>>(
        Xbf, Wih0bf, bih0, bhh0, xproj, M, N, K);
  }
  wmma_recurrence_kernel<<<RNN_NBLK, 256, 294912, stream>>>(
      xproj, Whh0bf, hbuf0, nullptr, ys0bf, h0T, cnts);

  // Layer 1: xproj1 = ys0 @ Wih1^T + bih1 + bhh1, then recurrence -> ys1 (fp32), h1T
  {
    const int M = T * B, N = H, K = H;
    const int blocks = (M / 128) * (N / 64);
    wmma_xproj_kernel<<<blocks, 256, (size_t)64 * K * sizeof(u16), stream>>>(
        ys0bf, Wih1bf, bih1, bhh1, xproj, M, N, K);
  }
  wmma_recurrence_kernel<<<RNN_NBLK, 256, 294912, stream>>>(
      xproj, Whh1bf, hbuf1, ys1, nullptr, h1T, cnts + T);
}
